// JiT_62139586838809
// MI455X (gfx1250) — matrix operand order repaired and checked by reading, not compiled, not run on hardware
//
#include <hip/hip_runtime.h>
#include <math.h>

// ---------------- model dims ----------------
#define BB 16
#define CC 3
#define IMGSZ 256
#define PP 16
#define NTOK 256          // patches per image
#define DD 1024
#define HH 16
#define HD 64
#define LL 8
#define PCAD 768
#define FHID 2730
#define FHPAD 2736        // padded stride for K=2730 operands (mult of 16)
#define FREQN 256
#define MROWS (BB * NTOK) // 4096 token rows

// ---------------- WMMA GEMM config ----------------
typedef __attribute__((ext_vector_type(16))) __bf16 v16bf;
typedef __attribute__((ext_vector_type(8)))  float  v8f;

#define BM 128
#define BN 128
#define BKK 32
#define LDS_STRIDE 40      // halves; 80B row stride -> 16B aligned, conflict-free b128
#define GEMM_THREADS 256

__device__ inline unsigned short f2bf(float f) {
  unsigned int u = __float_as_uint(f);
  u = u + 0x7FFFu + ((u >> 16) & 1u);   // round-to-nearest-even
  return (unsigned short)(u >> 16);
}

__device__ inline unsigned pack2bf(float a, float b) {
  return (unsigned)f2bf(a) | ((unsigned)f2bf(b) << 16);
}

__device__ inline v16bf mkfrag(uint4 lo, uint4 hi) {
  union { uint4 u[2]; v16bf v; } t;
  t.u[0] = lo; t.u[1] = hi;
  return t.v;
}

// A fragment 16x32 bf16: lane r=lane&15 holds row r; kh=lane>>4 selects K half.
__device__ inline v16bf load_frag_a(const unsigned short* As, int rowBase, int lane) {
  int r = lane & 15, kh = lane >> 4;
  const unsigned short* p = As + (rowBase + r) * LDS_STRIDE + kh * 8;
  uint4 lo = *(const uint4*)p;
  uint4 hi = *(const uint4*)(p + 16);
  return mkfrag(lo, hi);
}

// B fragment 32x16 bf16: lane n=lane&15 is column; kh=lane>>4 selects K half.
__device__ inline v16bf load_frag_b(const unsigned short* Bs, int colBase, int lane) {
  int n = lane & 15, kh = lane >> 4;
  const unsigned short* p = Bs + (colBase + n) * LDS_STRIDE + kh * 8;
  uint4 lo = *(const uint4*)p;
  uint4 hi = *(const uint4*)(p + 16);
  return mkfrag(lo, hi);
}

// async global->LDS copy of 16 bytes per lane (ASYNCcnt-tracked)
__device__ inline void async_b128(unsigned ldsAddr, const unsigned short* gaddr) {
  asm volatile("global_load_async_to_lds_b128 %0, %1, off"
               :: "v"(ldsAddr), "v"((unsigned long long)(uintptr_t)gaddr)
               : "memory");
}
__device__ inline void wait_async0() {
  asm volatile("s_wait_asynccnt 0x0" ::: "memory");
}

struct GemmP {
  const unsigned short* A; const unsigned short* B;   // bf16 operands
  const float* bias;
  float* Out;                 // optional fp32 output
  unsigned short* Outb;       // optional bf16 output (same layout/ldc)
  const float* res; const float* gate;
  int M, N, K, lda, ldb, ldc;
  float alpha;         // used by EPI=1 (scale+clip)
  int tpb, gldn;       // gate indexing: gate[(row/tpb)*gldn + col]
  int zdiv;            // batched pointer offsets: (z/zdiv)*s1 + (z%zdiv)*s2
  long long sA1, sA2, sB1, sB2, sC1, sC2;
};

// BTRANS=1: B is W[N,K] bf16 row-major -> A @ W^T.  BTRANS=0: B is Bmat[K,N] -> A @ Bmat.
// EPI: 0 = acc+bias ; 1 = clamp(acc*alpha, +-65504) ; 2 = res + gate*(acc+bias)
template<int BTRANS, int EPI>
__global__ __launch_bounds__(GEMM_THREADS)
void wmma_gemm(GemmP p) {
  __shared__ alignas(16) unsigned short As[2][BM * LDS_STRIDE];
  __shared__ alignas(16) unsigned short Bs[2][BN * LDS_STRIDE];

  const int z = blockIdx.z;
  const long long zq = z / p.zdiv, zr = z % p.zdiv;
  const unsigned short* A  = p.A + zq * p.sA1 + zr * p.sA2;
  const unsigned short* Bm = p.B + zq * p.sB1 + zr * p.sB2;
  const long long offC = zq * p.sC1 + zr * p.sC2;
  float*          Out  = p.Out  ? p.Out  + offC : nullptr;
  unsigned short* Outb = p.Outb ? p.Outb + offC : nullptr;

  const int tid  = threadIdx.x;
  const int m0   = blockIdx.y * BM;
  const int n0   = blockIdx.x * BN;
  const int lane = tid & 31;
  const int wave = tid >> 5;
  const int wm   = (wave >> 1) * 32;   // 4 wave-rows
  const int wn   = (wave & 1) * 64;    // 2 wave-cols

  const bool aFull = (m0 + BM <= p.M);
  const bool bFull = (n0 + BN <= p.N);

  // chunk coords: A / B(NT): 8 bf16 per chunk along K; two chunks rows r0 and r0+64
  const int r0  = tid >> 2;
  const int kc0 = (tid & 3) * 8;
  // B(NN): 8 bf16 per chunk along N; two chunks at k rows kk0 and kk0+16
  const int kk0 = tid >> 4;
  const int nc0 = (tid & 15) * 8;

  v8f acc[2][4];
  const v8f vzero = {0.f,0.f,0.f,0.f,0.f,0.f,0.f,0.f};
#pragma unroll
  for (int i = 0; i < 2; ++i)
#pragma unroll
    for (int j = 0; j < 4; ++j) acc[i][j] = vzero;

  union SR { uint4 v[2]; unsigned short s[16]; };

  // Issue the staging of K-slab k0 into LDS buffer `buf`.
  // Interior slabs: async global->LDS b128 (no VGPR round-trip).
  // Edge slabs / NN-transpose: guarded regs + ds_store.
  auto issueSlab = [&](int k0, int buf) {
    unsigned short* Ab = &As[buf][0];
    unsigned short* Bb = &Bs[buf][0];
    const bool kFull = (k0 + BKK <= p.K);
    // ---- A operand ----
    if (aFull && kFull) {
      const unsigned short* ga = A + (long long)(m0 + r0) * p.lda + k0 + kc0;
      async_b128((unsigned)(uintptr_t)&Ab[r0 * LDS_STRIDE + kc0], ga);
      async_b128((unsigned)(uintptr_t)&Ab[(r0 + 64) * LDS_STRIDE + kc0],
                 ga + (long long)64 * p.lda);
      if (k0 + 2 * BKK < p.K)
        __builtin_prefetch(ga + 2 * BKK, 0, 2);   // global_prefetch_b8
    } else {
      SR ar;
#pragma unroll
      for (int e = 0; e < 16; ++e) {
        int half = e >> 3, j = e & 7;
        int gm = m0 + r0 + half * 64;
        int gk = k0 + kc0 + j;
        ar.s[e] = (gm < p.M && gk < p.K) ? A[(long long)gm * p.lda + gk] : (unsigned short)0;
      }
      *(uint4*)(&Ab[r0 * LDS_STRIDE + kc0])        = ar.v[0];
      *(uint4*)(&Ab[(r0 + 64) * LDS_STRIDE + kc0]) = ar.v[1];
    }
    // ---- B operand ----
    if (BTRANS) {
      if (bFull && kFull) {
        const unsigned short* gb = Bm + (long long)(n0 + r0) * p.ldb + k0 + kc0;
        async_b128((unsigned)(uintptr_t)&Bb[r0 * LDS_STRIDE + kc0], gb);
        async_b128((unsigned)(uintptr_t)&Bb[(r0 + 64) * LDS_STRIDE + kc0],
                   gb + (long long)64 * p.ldb);
        if (k0 + 2 * BKK < p.K)
          __builtin_prefetch(gb + 2 * BKK, 0, 2);
      } else {
        SR br;
#pragma unroll
        for (int e = 0; e < 16; ++e) {
          int half = e >> 3, j = e & 7;
          int gn = n0 + r0 + half * 64;
          int gk = k0 + kc0 + j;
          br.s[e] = (gn < p.N && gk < p.K) ? Bm[(long long)gn * p.ldb + gk] : (unsigned short)0;
        }
        *(uint4*)(&Bb[r0 * LDS_STRIDE + kc0])        = br.v[0];
        *(uint4*)(&Bb[(r0 + 64) * LDS_STRIDE + kc0]) = br.v[1];
      }
    } else {
      // NN: must transpose into [n][k]; async cannot transpose -> sync path
      SR br;
      if (bFull && kFull) {
        const unsigned short* base = Bm + (long long)(k0 + kk0) * p.ldb + n0 + nc0;
        br.v[0] = *(const uint4*)base;
        br.v[1] = *(const uint4*)(base + (long long)16 * p.ldb);
      } else {
#pragma unroll
        for (int e = 0; e < 16; ++e) {
          int half = e >> 3, j = e & 7;
          int gk = k0 + kk0 + half * 16;
          int gn = n0 + nc0 + j;
          br.s[e] = (gk < p.K && gn < p.N) ? Bm[(long long)gk * p.ldb + gn] : (unsigned short)0;
        }
      }
#pragma unroll
      for (int j = 0; j < 8; ++j) {
        Bb[(nc0 + j) * LDS_STRIDE + kk0]      = br.s[j];
        Bb[(nc0 + j) * LDS_STRIDE + kk0 + 16] = br.s[8 + j];
      }
    }
  };

  // ---- double-buffered, async-staged main loop ----
  const int nIter = (p.K + BKK - 1) / BKK;
  issueSlab(0, 0);
  for (int it = 0; it < nIter; ++it) {
    const int cur = it & 1;
    wait_async0();        // this wave's async writes into buffer `cur` are done
    __syncthreads();      // all waves done writing `cur` AND done reading `cur^1`
    int kn = (it + 1) * BKK;
    if (kn < p.K) issueSlab(kn, cur ^ 1);   // overlap next-slab DMA with WMMA

    const unsigned short* Ab = &As[cur][0];
    const unsigned short* Bb = &Bs[cur][0];
    v16bf af[2], bfR[4];
#pragma unroll
    for (int i = 0; i < 2; ++i) af[i] = load_frag_a(Ab, wm + i * 16, lane);
#pragma unroll
    for (int j = 0; j < 4; ++j) bfR[j] = load_frag_b(Bb, wn + j * 16, lane);
#pragma unroll
    for (int i = 0; i < 2; ++i)
#pragma unroll
      for (int j = 0; j < 4; ++j)
        acc[i][j] = __builtin_amdgcn_wmma_f32_16x16x32_bf16(
            false, af[i], false, bfR[j], (short)0, acc[i][j], false, false);
  }

  // epilogue: lane = (m>=8)*16 + n ; vgpr r holds m = (lane>>4)*8 + r
  const int nIn = lane & 15;
  const int mh  = lane >> 4;
#pragma unroll
  for (int i = 0; i < 2; ++i) {
#pragma unroll
    for (int j = 0; j < 4; ++j) {
      int gcol = n0 + wn + j * 16 + nIn;
      if (gcol >= p.N) continue;
      float bv = p.bias ? p.bias[gcol] : 0.f;
#pragma unroll
      for (int r = 0; r < 8; ++r) {
        int grow = m0 + wm + i * 16 + mh * 8 + r;
        if (grow >= p.M) continue;
        float v = acc[i][j][r];
        if (EPI == 0) {
          v = v + bv;
        } else if (EPI == 1) {
          v = fminf(fmaxf(v * p.alpha, -65504.f), 65504.f);
        } else {
          float g = p.gate[(grow / p.tpb) * p.gldn + gcol];
          v = p.res[(long long)grow * p.ldc + gcol] + g * (v + bv);
        }
        long long oi = (long long)grow * p.ldc + gcol;
        if (Out)  Out[oi]  = v;
        if (Outb) Outb[oi] = f2bf(v);
      }
    }
  }
}

static GemmP mk_gemm(const unsigned short* A, const unsigned short* B, const float* bias,
                     float* Out, unsigned short* Outb,
                     int M, int N, int K, int lda, int ldb, int ldc) {
  GemmP p;
  p.A = A; p.B = B; p.bias = bias; p.Out = Out; p.Outb = Outb;
  p.res = nullptr; p.gate = nullptr;
  p.M = M; p.N = N; p.K = K; p.lda = lda; p.ldb = ldb; p.ldc = ldc;
  p.alpha = 1.f; p.tpb = NTOK; p.gldn = 0;
  p.zdiv = 1; p.sA1 = p.sA2 = p.sB1 = p.sB2 = p.sC1 = p.sC2 = 0;
  return p;
}

template<int BT, int EPI>
static void run_gemm(const GemmP& p, int batch, hipStream_t s) {
  dim3 g((p.N + BN - 1) / BN, (p.M + BM - 1) / BM, batch);
  wmma_gemm<BT, EPI><<<g, GEMM_THREADS, 0, s>>>(p);
}

// ---------------- conversion / elementwise kernels ----------------

// fp32 [rows x K, ldsrc] -> bf16 [rows x K, lddst]   (K even; strides even)
__global__ void cvt2d_k(const float* __restrict__ in, unsigned short* __restrict__ outp,
                        int rows, int K, int ldsrc, int lddst) {
  long long c = (long long)blockIdx.x * 256 + threadIdx.x;
  int kh = K >> 1;
  long long tot = (long long)rows * kh;
  if (c >= tot) return;
  int r = (int)(c / kh);
  int k = (int)(c % kh) * 2;
  float2 v = *(const float2*)(in + (long long)r * ldsrc + k);
  *(unsigned*)(outp + (long long)r * lddst + k) = pack2bf(v.x, v.y);
}

// x [B,C,256,256] -> bf16 patches [B*256, 768]
__global__ void patchify_k(const float* __restrict__ x, unsigned short* __restrict__ out) {
  int idx = blockIdx.x * 256 + threadIdx.x;
  if (idx >= MROWS * (CC * PP * PP)) return;
  int col = idx % 768, row = idx / 768;
  int b = row >> 8, pidx = row & 255;
  int hy = pidx >> 4, hx = pidx & 15;
  int c = col / 256, rem = col % 256;
  int py = rem >> 4, px = rem & 15;
  out[idx] = f2bf(x[(((long long)b * CC + c) * IMGSZ + hy * PP + py) * IMGSZ + hx * PP + px]);
}

// timestep embedding [B, 256]
__global__ void tembed_k(const float* __restrict__ t, float* __restrict__ te) {
  int idx = blockIdx.x * 256 + threadIdx.x;
  if (idx >= BB * FREQN) return;
  int b = idx >> 8, i = idx & 255;
  int j = i & 127;
  float fr = expf(-logf(10000.f) * (float)j / 128.f);
  float a = t[b] * fr;
  te[idx] = (i < 128) ? cosf(a) : sinf(a);
}

// naive small GEMM for conditioning path (fp32): out[m,n] = act(A@W^T + bias)
__global__ void sgemm_k(const float* __restrict__ A, const float* __restrict__ W,
                        const float* __restrict__ bias, float* __restrict__ out,
                        int M, int N, int K, int act) {
  int idx = blockIdx.x * 256 + threadIdx.x;
  if (idx >= M * N) return;
  int m = idx / N, n = idx % N;
  float s = bias ? bias[n] : 0.f;
  const float* a = A + (long long)m * K;
  const float* w = W + (long long)n * K;
  for (int k = 0; k < K; ++k) s += a[k] * w[k];
  if (act == 1) s = s / (1.f + expf(-s));   // SiLU
  out[idx] = s;
}

// c_act = silu(ct + label_tab[y])
__global__ void cond_final_k(const float* __restrict__ ct, const float* __restrict__ tab,
                             const int* __restrict__ y, float* __restrict__ cact) {
  int idx = blockIdx.x * 256 + threadIdx.x;
  if (idx >= BB * DD) return;
  int b = idx >> 10, d = idx & 1023;
  float v = ct[idx] + tab[(long long)y[b] * DD + d];
  cact[idx] = v / (1.f + expf(-v));
}

// modulated RMSNorm -> bf16: out = rms(x)*nw*(1+sc) + sh ; one block per token row
__global__ __launch_bounds__(256)
void normmod_k(const float* __restrict__ X, const float* __restrict__ nw,
               const float* __restrict__ mod, int ldmod, int off_sh, int off_sc,
               unsigned short* __restrict__ out, int tpb) {
  int row = blockIdx.x;
  int tid = threadIdx.x;
  const float* xr = X + (long long)row * DD;
  float4 v = ((const float4*)xr)[tid];
  float ss = v.x * v.x + v.y * v.y + v.z * v.z + v.w * v.w;
  __shared__ float red[256];
  red[tid] = ss; __syncthreads();
  for (int s = 128; s > 0; s >>= 1) { if (tid < s) red[tid] += red[tid + s]; __syncthreads(); }
  float r = rsqrtf(red[0] / (float)DD + 1e-6f);
  int b = row / tpb;
  const float* sh = mod + (long long)b * ldmod + off_sh;
  const float* sc = mod + (long long)b * ldmod + off_sc;
  int d0 = tid * 4;
  float xv[4] = {v.x, v.y, v.z, v.w};
  float yv[4];
#pragma unroll
  for (int i = 0; i < 4; ++i) {
    int d = d0 + i;
    yv[i] = xv[i] * r * nw[d] * (1.f + sc[d]) + sh[d];
  }
  uint2 pk; pk.x = pack2bf(yv[0], yv[1]); pk.y = pack2bf(yv[2], yv[3]);
  *(uint2*)(out + (long long)row * DD + d0) = pk;
}

// per-head RMSNorm + RoPE on q,k: read fp32 qkv, write bf16 q/k slices of qkv_bf
__global__ void qk_prep_k(const float* __restrict__ qkvf, unsigned short* __restrict__ qkvbf,
                          const float* __restrict__ qnw, const float* __restrict__ knw) {
  int idx = blockIdx.x * 256 + threadIdx.x;
  if (idx >= 2 * BB * HH * NTOK) return;
  int which = idx >> 16;            // 0 = q, 1 = k
  int rest  = idx & 65535;
  int b = rest >> 12;
  int h = (rest >> 8) & 15;
  int n = rest & 255;
  long long base = ((long long)(b * NTOK + n)) * (3 * DD) + which * DD + h * HD;
  const float* pf = qkvf + base;
  unsigned short* pb = qkvbf + base;
  const float* w = which ? knw : qnw;
  float v[HD]; float ss = 0.f;
#pragma unroll
  for (int d = 0; d < HD; ++d) { v[d] = pf[d]; ss += v[d] * v[d]; }
  float r = rsqrtf(ss / (float)HD + 1e-6f);
#pragma unroll
  for (int d = 0; d < HD / 2; ++d) {
    float x1 = v[d] * r * w[d];
    float x2 = v[d + 32] * r * w[d + 32];
    float inv = expf(-logf(10000.f) * (float)d / 32.f);
    float ang = (float)n * inv;
    float c = cosf(ang), s = sinf(ang);
    pb[d]      = f2bf(x1 * c - x2 * s);
    pb[d + 32] = f2bf(x2 * c + x1 * s);
  }
}

// row softmax (fp32 in, bf16 out), 256 cols per row
__global__ __launch_bounds__(256)
void softmax_k(const float* __restrict__ S, unsigned short* __restrict__ Sb) {
  long long row = blockIdx.x;
  int tid = threadIdx.x;
  const float* p = S + row * NTOK;
  float v = p[tid];
  __shared__ float red[256];
  red[tid] = v; __syncthreads();
  for (int s = 128; s > 0; s >>= 1) { if (tid < s) red[tid] = fmaxf(red[tid], red[tid + s]); __syncthreads(); }
  float mx = red[0]; __syncthreads();
  float e = expf(v - mx);
  red[tid] = e; __syncthreads();
  for (int s = 128; s > 0; s >>= 1) { if (tid < s) red[tid] += red[tid + s]; __syncthreads(); }
  Sb[row * NTOK + tid] = f2bf(e / red[0]);
}

// hid(bf16, padded ld) = silu(x12[:, :FH]) * x12[:, FH:]
__global__ void swiglu_k(const float* __restrict__ x12, unsigned short* __restrict__ hid) {
  long long idx = (long long)blockIdx.x * 256 + threadIdx.x;
  if (idx >= (long long)MROWS * FHID) return;
  long long m = idx / FHID; int j = (int)(idx % FHID);
  float x1 = x12[m * (2 * FHID) + j];
  float x2 = x12[m * (2 * FHID) + FHID + j];
  hid[m * FHPAD + j] = f2bf(x1 / (1.f + expf(-x1)) * x2);
}

// outp [B*256, 768] -> out [B,C,256,256]
__global__ void unpatch_k(const float* __restrict__ op, float* __restrict__ out) {
  int idx = blockIdx.x * 256 + threadIdx.x;
  if (idx >= BB * CC * IMGSZ * IMGSZ) return;
  int w = idx & 255;
  int hgt = (idx >> 8) & 255;
  int rest = idx >> 16;
  int c = rest % CC, b = rest / CC;
  int hy = hgt >> 4, py = hgt & 15, hx = w >> 4, px = w & 15;
  out[idx] = op[((long long)(b * NTOK + hy * PP + hx)) * 768 + (py * PP + px) * CC + c];
}

// ---------------- launcher ----------------
static void cvt_w(const float* src, unsigned short* dst, int rows, int K, int ldsrc, int lddst,
                  hipStream_t s) {
  long long tot = (long long)rows * (K >> 1);
  cvt2d_k<<<(int)((tot + 255) / 256), 256, 0, s>>>(src, dst, rows, K, ldsrc, lddst);
}

extern "C" void kernel_launch(void* const* d_in, const int* in_sizes, int n_in,
                              void* d_out, int out_size, void* d_ws, size_t ws_size,
                              hipStream_t stream) {
  (void)in_sizes; (void)n_in; (void)out_size; (void)ws_size;
  const float* x        = (const float*)d_in[0];
  const float* t        = (const float*)d_in[1];
  const int*   y        = (const int*)  d_in[2];
  const float* pe1_w    = (const float*)d_in[3];
  const float* pe2_w    = (const float*)d_in[4];
  const float* pe2_b    = (const float*)d_in[5];
  const float* te_w1    = (const float*)d_in[6];
  const float* te_b1    = (const float*)d_in[7];
  const float* te_w2    = (const float*)d_in[8];
  const float* te_b2    = (const float*)d_in[9];
  const float* labelTab = (const float*)d_in[10];
  const float* qkv_w    = (const float*)d_in[11];
  const float* qkv_b    = (const float*)d_in[12];
  const float* proj_w   = (const float*)d_in[13];
  const float* proj_b   = (const float*)d_in[14];
  const float* qn_w     = (const float*)d_in[15];
  const float* kn_w     = (const float*)d_in[16];
  const float* n1_w     = (const float*)d_in[17];
  const float* n2_w     = (const float*)d_in[18];
  const float* ada_w    = (const float*)d_in[19];
  const float* ada_b    = (const float*)d_in[20];
  const float* w12_w    = (const float*)d_in[21];
  const float* w12_b    = (const float*)d_in[22];
  const float* w3_w     = (const float*)d_in[23];
  const float* w3_b     = (const float*)d_in[24];
  const float* flada_w  = (const float*)d_in[25];
  const float* flada_b  = (const float*)d_in[26];
  const float* flnorm_w = (const float*)d_in[27];
  const float* fllin_w  = (const float*)d_in[28];
  const float* fllin_b  = (const float*)d_in[29];
  float* out = (float*)d_out;

  // ---- workspace carve-up ----
  float* ws = (float*)d_ws;
  size_t off = 0;
  auto alloc = [&](size_t n) { float* p = ws + off; off += n; return p; };
  float* hx      = alloc((size_t)MROWS * DD);           // fp32 residual stream
  float* qkvb    = alloc((size_t)MROWS * 3 * DD);       // fp32 qkv (pre-RMS/RoPE)
  float* scoresb = alloc((size_t)BB * HH * NTOK * NTOK);// fp32 logits
  float* x12     = alloc((size_t)MROWS * 2 * FHID);     // fp32 SwiGLU pre-act
  float* outp    = alloc((size_t)MROWS * 768);
  float* te      = alloc((size_t)BB * FREQN);
  float* th1     = alloc((size_t)BB * DD);
  float* ct      = alloc((size_t)BB * DD);
  float* cact    = alloc((size_t)BB * DD);
  float* modb    = alloc((size_t)BB * 6 * DD);
  float* fmodb   = alloc((size_t)BB * 2 * DD);

  unsigned short* wsu = (unsigned short*)(ws + off);
  size_t uoff = 0;
  auto ualloc = [&](size_t n) { unsigned short* p = wsu + uoff; uoff += n; return p; };
  unsigned short* patchesbf = ualloc((size_t)MROWS * 768);
  unsigned short* hbufbf    = ualloc((size_t)MROWS * DD);     // normed activations
  unsigned short* qkvbf     = ualloc((size_t)MROWS * 3 * DD);
  unsigned short* scoresbf  = ualloc((size_t)BB * HH * NTOK * NTOK);
  unsigned short* obufbf    = ualloc((size_t)MROWS * DD);
  unsigned short* hidbf     = ualloc((size_t)MROWS * FHPAD);
  unsigned short* wbf       = ualloc((size_t)2 * FHID * DD);  // weight scratch (max = w12)

  // ---- patch embed ----
  patchify_k<<<(MROWS * 768 + 255) / 256, 256, 0, stream>>>(x, patchesbf);
  cvt_w(pe1_w, wbf, PCAD, 768, 768, 768, stream);
  { GemmP p = mk_gemm(patchesbf, wbf, nullptr, nullptr, hbufbf, MROWS, PCAD, 768, 768, 768, PCAD);
    run_gemm<1, 0>(p, 1, stream); }
  cvt_w(pe2_w, wbf, DD, PCAD, PCAD, PCAD, stream);
  { GemmP p = mk_gemm(hbufbf, wbf, pe2_b, hx, nullptr, MROWS, DD, PCAD, PCAD, PCAD, DD);
    run_gemm<1, 0>(p, 1, stream); }

  // ---- conditioning (tiny, fp32) ----
  tembed_k<<<(BB * FREQN + 255) / 256, 256, 0, stream>>>(t, te);
  sgemm_k<<<(BB * DD + 255) / 256, 256, 0, stream>>>(te, te_w1, te_b1, th1, BB, DD, FREQN, 1);
  sgemm_k<<<(BB * DD + 255) / 256, 256, 0, stream>>>(th1, te_w2, te_b2, ct, BB, DD, DD, 0);
  cond_final_k<<<(BB * DD + 255) / 256, 256, 0, stream>>>(ct, labelTab, y, cact);

  // ---- transformer blocks ----
  for (int l = 0; l < LL; ++l) {
    const float* qw   = qkv_w  + (size_t)l * 3 * DD * DD;
    const float* qb   = qkv_b  + (size_t)l * 3 * DD;
    const float* pw   = proj_w + (size_t)l * DD * DD;
    const float* pb   = proj_b + (size_t)l * DD;
    const float* qnw  = qn_w   + (size_t)l * HD;
    const float* knw  = kn_w   + (size_t)l * HD;
    const float* n1w  = n1_w   + (size_t)l * DD;
    const float* n2w  = n2_w   + (size_t)l * DD;
    const float* aw   = ada_w  + (size_t)l * 6 * DD * DD;
    const float* ab   = ada_b  + (size_t)l * 6 * DD;
    const float* w12w = w12_w  + (size_t)l * 2 * FHID * DD;
    const float* w12b = w12_b  + (size_t)l * 2 * FHID;
    const float* w3w  = w3_w   + (size_t)l * DD * FHID;
    const float* w3b  = w3_b   + (size_t)l * DD;

    // adaLN modulation [16, 6144]
    sgemm_k<<<(BB * 6 * DD + 255) / 256, 256, 0, stream>>>(cact, aw, ab, modb, BB, 6 * DD, DD, 0);

    // attention branch
    normmod_k<<<MROWS, 256, 0, stream>>>(hx, n1w, modb, 6 * DD, 0, DD, hbufbf, NTOK);
    cvt_w(qw, wbf, 3 * DD, DD, DD, DD, stream);
    { GemmP p = mk_gemm(hbufbf, wbf, qb, qkvb, qkvbf, MROWS, 3 * DD, DD, DD, DD, 3 * DD);
      run_gemm<1, 0>(p, 1, stream); }
    qk_prep_k<<<(2 * BB * HH * NTOK + 255) / 256, 256, 0, stream>>>(qkvb, qkvbf, qnw, knw);

    // scores = clip(q @ k^T / 8), batched over B*H heads
    { GemmP p = mk_gemm(qkvbf, qkvbf + DD, nullptr, scoresb, nullptr,
                        NTOK, NTOK, HD, 3 * DD, 3 * DD, NTOK);
      p.alpha = 0.125f; p.zdiv = HH;
      p.sA1 = (long long)NTOK * 3 * DD; p.sA2 = HD;
      p.sB1 = (long long)NTOK * 3 * DD; p.sB2 = HD;
      p.sC1 = (long long)HH * NTOK * NTOK; p.sC2 = (long long)NTOK * NTOK;
      run_gemm<1, 1>(p, BB * HH, stream); }
    softmax_k<<<BB * HH * NTOK, 256, 0, stream>>>(scoresb, scoresbf);
    // o = attn @ v (NN) -> bf16 [B, N, H, HD]
    { GemmP p = mk_gemm(scoresbf, qkvbf + 2 * DD, nullptr, nullptr, obufbf,
                        NTOK, HD, NTOK, NTOK, 3 * DD, DD);
      p.zdiv = HH;
      p.sA1 = (long long)HH * NTOK * NTOK; p.sA2 = (long long)NTOK * NTOK;
      p.sB1 = (long long)NTOK * 3 * DD; p.sB2 = HD;
      p.sC1 = (long long)NTOK * DD; p.sC2 = HD;
      run_gemm<0, 0>(p, BB * HH, stream); }
    // proj + gated residual: hx = hx + g_a * (o @ pw^T + pb)
    cvt_w(pw, wbf, DD, DD, DD, DD, stream);
    { GemmP p = mk_gemm(obufbf, wbf, pb, hx, nullptr, MROWS, DD, DD, DD, DD, DD);
      p.res = hx; p.gate = modb + 2 * DD; p.tpb = NTOK; p.gldn = 6 * DD;
      run_gemm<1, 2>(p, 1, stream); }

    // SwiGLU MLP branch
    normmod_k<<<MROWS, 256, 0, stream>>>(hx, n2w, modb, 6 * DD, 3 * DD, 4 * DD, hbufbf, NTOK);
    cvt_w(w12w, wbf, 2 * FHID, DD, DD, DD, stream);
    { GemmP p = mk_gemm(hbufbf, wbf, w12b, x12, nullptr, MROWS, 2 * FHID, DD, DD, DD, 2 * FHID);
      run_gemm<1, 0>(p, 1, stream); }
    { long long tot = (long long)MROWS * FHID;
      swiglu_k<<<(int)((tot + 255) / 256), 256, 0, stream>>>(x12, hidbf); }
    cvt_w(w3w, wbf, DD, FHID, FHID, FHPAD, stream);
    { GemmP p = mk_gemm(hidbf, wbf, w3b, hx, nullptr, MROWS, DD, FHID, FHPAD, FHPAD, DD);
      p.res = hx; p.gate = modb + 5 * DD; p.tpb = NTOK; p.gldn = 6 * DD;
      run_gemm<1, 2>(p, 1, stream); }
  }

  // ---- final layer ----
  sgemm_k<<<(BB * 2 * DD + 255) / 256, 256, 0, stream>>>(cact, flada_w, flada_b, fmodb, BB, 2 * DD, DD, 0);
  normmod_k<<<MROWS, 256, 0, stream>>>(hx, flnorm_w, fmodb, 2 * DD, 0, DD, hbufbf, NTOK);
  cvt_w(fllin_w, wbf, 768, DD, DD, DD, stream);
  { GemmP p = mk_gemm(hbufbf, wbf, fllin_b, outp, nullptr, MROWS, 768, DD, DD, DD, 768);
    run_gemm<1, 0>(p, 1, stream); }
  unpatch_k<<<(BB * CC * IMGSZ * IMGSZ + 255) / 256, 256, 0, stream>>>(outp, out);
}
